// C2LIP_Loss_83648783057700
// MI455X (gfx1250) — matrix operand order repaired and checked by reading, not compiled, not run on hardware
//
#include <hip/hip_runtime.h>

// ---------------- problem constants ----------------
#define Bv   128
#define Lv   577
#define Dv   768
#define NPv  1024
#define LTv  37     // ceil(L/16)
#define LPv  592    // LTv*16 (padded L)
#define SMOOTHv 4.0f
#define EPSv    1e-8f

typedef __attribute__((ext_vector_type(16))) __bf16 v16bf;
typedef __attribute__((ext_vector_type(8)))  __bf16 v8bf;
typedef __attribute__((ext_vector_type(8)))  float  v8f;

// cat two v8bf into v16bf
__device__ inline v16bf cat8(v8bf lo, v8bf hi) {
    return __builtin_shufflevector(lo, hi, 0,1,2,3,4,5,6,7,8,9,10,11,12,13,14,15);
}

// A fragment (16M x 32K bf16). rowk = pointer to this lane's row (M=lane&15) at K-offset.
// Lanes 0-15 hold K {0..7, 16..23}; lanes 16-31 hold K {8..15, 24..31}.
__device__ inline v16bf fragA_row(const __bf16* rowk) {
    int half = (threadIdx.x >> 4) & 1;
    v8bf lo = *(const v8bf*)(rowk + half*8);
    v8bf hi = *(const v8bf*)(rowk + half*8 + 16);
    return cat8(lo, hi);
}

// B fragment (32K x 16N bf16), stored as B^T rows: rowk points to this lane's
// column (N=lane&15) laid out contiguously over K. Lanes 0-15: K {0..7, 16..23};
// lanes 16-31: K {8..15, 24..31}.
__device__ inline v16bf fragB_row(const __bf16* rowk) {
    int half = (threadIdx.x >> 4) & 1;
    v8bf lo = *(const v8bf*)(rowk + half*8);
    v8bf hi = *(const v8bf*)(rowk + half*8 + 16);
    return cat8(lo, hi);
}

__device__ inline v8f wmma_bf16(v16bf a, v16bf b, v8f c) {
    return __builtin_amdgcn_wmma_f32_16x16x32_bf16(
        /*neg_a=*/false, a, /*neg_b=*/false, b,
        /*c_mod=*/(short)0, c, /*reuse_a=*/false, /*reuse_b=*/false);
}

// -log(sigmoid(x)), numerically stable
__device__ inline float neg_logsig(float x) {
    return log1pf(__expf(-fabsf(x))) - fminf(x, 0.0f);
}

// ---------------- K0: fp32 -> bf16 convert ----------------
__global__ __launch_bounds__(256) void k0_convert(const float* __restrict__ src,
                                                  __bf16* __restrict__ dst, long n) {
    long i = (long)blockIdx.x * blockDim.x + threadIdx.x;
    long stride = (long)gridDim.x * blockDim.x;
    for (; i < n; i += stride) dst[i] = (__bf16)src[i];
}

// ---------------- K1: rownorm[b,l] = max(||leaky(attn[b,l,:])||_2, eps) ----------------
// grid (LTv, Bv), block 256 (8 waves). Each block: one l-tile of 16 rows, sweep all NP.
__global__ __launch_bounds__(256) void k1_rownorm(const __bf16* __restrict__ tokb,
                                                  const __bf16* __restrict__ npb,
                                                  float* __restrict__ rownorm) {
    __shared__ __align__(16) __bf16 tokT[16][Dv];
    __shared__ float rowss[16];
    const int b = blockIdx.y, l0 = blockIdx.x * 16, tid = threadIdx.x;
    if (tid < 16) rowss[tid] = 0.0f;
    // stage l-tile (clamped rows; pad rows produce unread garbage norms)
    for (int i = tid; i < 16 * (Dv/8); i += 256) {
        int r = i / (Dv/8), c8 = i % (Dv/8);
        int gl = l0 + r; if (gl > Lv-1) gl = Lv-1;
        *(v8bf*)&tokT[r][c8*8] = *(const v8bf*)(tokb + ((size_t)b*Lv + gl)*Dv + c8*8);
    }
    __syncthreads();
    const int wave = tid >> 5, lane = tid & 31, half = lane >> 4, l16 = lane & 15;
    float ss[8];
#pragma unroll
    for (int r = 0; r < 8; r++) ss[r] = 0.0f;
    for (int nt = wave; nt < NPv/16; nt += 8) {
        v8f c = (v8f)0.0f;
        const __bf16* arow = &tokT[l16][0];
        const __bf16* brow = npb + (size_t)(nt*16 + l16) * Dv;
        for (int k = 0; k < Dv; k += 32) {
            v16bf a  = fragA_row(arow + k);
            v16bf bb = fragB_row(brow + k);
            c = wmma_bf16(a, bb, c);
        }
#pragma unroll
        for (int r = 0; r < 8; r++) {
            float x = c[r];
            x = (x >= 0.0f) ? x : 0.1f * x;   // leaky_relu(0.1)
            ss[r] += x * x;
        }
    }
#pragma unroll
    for (int r = 0; r < 8; r++) atomicAdd(&rowss[r + 8*half], ss[r]);
    __syncthreads();
    if (tid < 16) {
        int gl = l0 + tid;
        if (gl < Lv) rownorm[b*Lv + gl] = fmaxf(sqrtf(rowss[tid]), EPSv);
    }
}

// ---------------- K2: colmax/colsum[b,n] of SMOOTH*attn_norm over l ----------------
// grid (NPv/16, Bv), block 256.
__global__ __launch_bounds__(256) void k2_colstats(const __bf16* __restrict__ tokb,
                                                   const __bf16* __restrict__ npb,
                                                   const float* __restrict__ rownorm,
                                                   float* __restrict__ colmaxG,
                                                   float* __restrict__ colsumG) {
    __shared__ __align__(16) __bf16 npA[16][Dv];
    __shared__ float strip[16][LPv];
    __shared__ float pred[16][16];
    __shared__ float cmax[16];
    const int b = blockIdx.y, n0 = blockIdx.x * 16, tid = threadIdx.x;
    for (int i = tid; i < 16 * (Dv/8); i += 256) {
        int r = i / (Dv/8), c8 = i % (Dv/8);
        *(v8bf*)&npA[r][c8*8] = *(const v8bf*)(npb + (size_t)(n0 + r)*Dv + c8*8);
    }
    __syncthreads();
    const int wave = tid >> 5, lane = tid & 31, half = lane >> 4, l16 = lane & 15;
    for (int lt = wave; lt < LTv; lt += 8) {
        int lcol = lt*16 + l16;
        int gl = (lcol > Lv-1) ? (Lv-1) : lcol;
        float rn = rownorm[b*Lv + gl];
        v8f c = (v8f)0.0f;
        const __bf16* arow = &npA[l16][0];
        const __bf16* brow = tokb + ((size_t)b*Lv + gl)*Dv;
        for (int k = 0; k < Dv; k += 32) {
            v16bf a  = fragA_row(arow + k);
            v16bf bb = fragB_row(brow + k);
            c = wmma_bf16(a, bb, c);
        }
#pragma unroll
        for (int r = 0; r < 8; r++) {
            float x = c[r];
            x = (x >= 0.0f) ? x : 0.1f * x;
            float aS = (lcol < Lv) ? (SMOOTHv * x / rn) : -1e30f;
            strip[r + 8*half][lcol] = aS;
        }
    }
    __syncthreads();
    // pass 1: max over l for each n
    const int n = tid >> 4, ls = tid & 15;
    float m = -1e30f;
    for (int l = ls; l < LPv; l += 16) m = fmaxf(m, strip[n][l]);
    pred[n][ls] = m;
    __syncthreads();
    if (tid < 16) {
        float mm = -1e30f;
        for (int j = 0; j < 16; j++) mm = fmaxf(mm, pred[tid][j]);
        cmax[tid] = mm;
    }
    __syncthreads();
    // pass 2: sum of exp
    float s = 0.0f;
    for (int l = ls; l < LPv; l += 16) s += __expf(strip[n][l] - cmax[n]);
    __syncthreads();   // pred reuse
    pred[n][ls] = s;
    __syncthreads();
    if (tid < 16) {
        float ssum = 0.0f;
        for (int j = 0; j < 16; j++) ssum += pred[tid][j];
        colmaxG[b*NPv + n0 + tid] = cmax[tid];
        colsumG[b*NPv + n0 + tid] = ssum;
    }
}

// ---------------- K3: fused softmax-weights + second GEMM + cosine/XAC loss ----------------
// grid (NPv/16, Bv), block 256 (8 waves). Each wave owns 96 d-columns of wctx.
__global__ __launch_bounds__(256) void k3_xac(const __bf16* __restrict__ tokb,
                                              const __bf16* __restrict__ npb,
                                              const float* __restrict__ npf,
                                              const float* __restrict__ rownorm,
                                              const float* __restrict__ colmaxG,
                                              const float* __restrict__ colsumG,
                                              const int* __restrict__ npidx,
                                              const float* __restrict__ scaleP,
                                              const float* __restrict__ biasP,
                                              float* __restrict__ out) {
    __shared__ __align__(16) __bf16 npA[16][Dv];
    __shared__ __align__(16) __bf16 tokTr[Dv][32];   // transposed chunk, pad cols zeroed
    __shared__ float attnF[16][32];
    __shared__ __align__(16) __bf16 wT[16][32];
    __shared__ float cm[16], cs[16], numL[16], ssL[16];
    const int b = blockIdx.y, n0 = blockIdx.x * 16, tid = threadIdx.x;
    if (tid < 16) {
        cm[tid] = colmaxG[b*NPv + n0 + tid];
        cs[tid] = colsumG[b*NPv + n0 + tid];
        numL[tid] = 0.0f; ssL[tid] = 0.0f;
    }
    for (int i = tid; i < 16 * (Dv/8); i += 256) {
        int r = i / (Dv/8), c8 = i % (Dv/8);
        *(v8bf*)&npA[r][c8*8] = *(const v8bf*)(npb + (size_t)(n0 + r)*Dv + c8*8);
    }
    const int wave = tid >> 5, lane = tid & 31, half = lane >> 4, l16 = lane & 15;
    const int dwave = wave * 96;
    v8f acc[6];
#pragma unroll
    for (int dt = 0; dt < 6; dt++) acc[dt] = (v8f)0.0f;

    for (int cch = 0; cch < 19; cch++) {       // 19*32 = 608 >= 577
        const int l0 = cch * 32;
        __syncthreads();                        // protect LDS reuse across chunks
        for (int i = tid; i < 512; i += 256) ((float*)attnF)[i] = 0.0f;
        // stage transposed token chunk; zero pad columns (=> pad rows contribute 0)
        for (int i = tid; i < 32 * (Dv/8); i += 256) {
            int l = i / (Dv/8), c8 = i % (Dv/8);
            int gl = l0 + l;
            if (gl < Lv) {
                v8bf v = *(const v8bf*)(tokb + ((size_t)b*Lv + gl)*Dv + c8*8);
#pragma unroll
                for (int j = 0; j < 8; j++) tokTr[c8*8 + j][l] = v[j];
            } else {
#pragma unroll
                for (int j = 0; j < 8; j++) tokTr[c8*8 + j][l] = (__bf16)0.0f;
            }
        }
        __syncthreads();
        // GEMM1 partial: attnT[n, l-chunk] ; l-half = wave&1, K-slice = (wave>>1)*192
        {
            const int lh = wave & 1, kb = (wave >> 1) * 192;
            int lcol = l0 + lh*16 + l16;
            int gl = (lcol > Lv-1) ? (Lv-1) : lcol;
            v8f c = (v8f)0.0f;
            const __bf16* arow = &npA[l16][0];
            const __bf16* brow = tokb + ((size_t)b*Lv + gl)*Dv;
#pragma unroll
            for (int kk = 0; kk < 192; kk += 32) {
                v16bf a  = fragA_row(arow + kb + kk);
                v16bf bb = fragB_row(brow + kb + kk);
                c = wmma_bf16(a, bb, c);
            }
#pragma unroll
            for (int r = 0; r < 8; r++)
                atomicAdd(&attnF[r + 8*half][lh*16 + l16], c[r]);
        }
        __syncthreads();
        // transform: leaky -> /rownorm -> softmax weight (bf16)
        for (int i = tid; i < 512; i += 256) {
            int n = i >> 5, l = i & 31, gl = l0 + l;
            float wv = 0.0f;
            if (gl < Lv) {
                float x = attnF[n][l];
                x = (x >= 0.0f) ? x : 0.1f * x;
                float aS = SMOOTHv * x / rownorm[b*Lv + gl];
                wv = __expf(aS - cm[n]) / cs[n];
            }
            wT[n][l] = (__bf16)wv;
        }
        __syncthreads();
        // GEMM2: wctx[n, d-cols of this wave] += w @ tok
        {
            v16bf a = fragA_row(&wT[l16][0]);
#pragma unroll
            for (int dt = 0; dt < 6; dt++) {
                v16bf bb = fragB_row(&tokTr[dwave + dt*16 + l16][0]);
                acc[dt] = wmma_bf16(a, bb, acc[dt]);
            }
        }
    }
    // epilogue: num = sum_d q*wctx, ss = sum_d wctx^2 per n-row
    float nump[8], ssp[8];
#pragma unroll
    for (int r = 0; r < 8; r++) { nump[r] = 0.0f; ssp[r] = 0.0f; }
#pragma unroll
    for (int dt = 0; dt < 6; dt++) {
        int d = dwave + dt*16 + l16;
#pragma unroll
        for (int r = 0; r < 8; r++) {
            int n = n0 + r + 8*half;
            float wv = acc[dt][r];
            float qv = npf[(size_t)n*Dv + d];
            nump[r] += qv * wv;
            ssp[r]  += wv * wv;
        }
    }
#pragma unroll
    for (int r = 0; r < 8; r++) {
        atomicAdd(&numL[r + 8*half], nump[r]);
        atomicAdd(&ssL [r + 8*half], ssp[r]);
    }
    __syncthreads();
    if (tid < 16) {
        int n = n0 + tid;
        float qq = 0.0f;
        for (int d = 0; d < Dv; d++) { float q = npf[(size_t)n*Dv + d]; qq += q*q; }
        float den = fmaxf(sqrtf(qq) * sqrtf(ssL[tid]), EPSv);
        float sim = numL[tid] / den;
        float z = sim * scaleP[0] + biasP[0];
        float lbl = (npidx[n] == b) ? 1.0f : -1.0f;
        atomicAdd(out, neg_logsig(lbl * z) * (0.01f / (float)NPv));   // XAC_SCALE/NP
    }
}

// ---------------- K4: contrastive + NPC (tiny) ----------------
__global__ __launch_bounds__(256) void k4_sig(const float* __restrict__ img,
                                              const float* __restrict__ txt,
                                              const float* __restrict__ npf,
                                              const int* __restrict__ npidx,
                                              const float* __restrict__ scaleP,
                                              const float* __restrict__ biasP,
                                              float* __restrict__ out) {
    __shared__ float qi[Dv];
    __shared__ float red[256];
    const int i = blockIdx.x, tid = threadIdx.x;
    for (int d = tid; d < Dv; d += 256) qi[d] = img[(size_t)i*Dv + d];
    __syncthreads();
    const float sc = scaleP[0], bi = biasP[0];
    float local = 0.0f;
    if (tid < Bv) {
        float dot = 0.0f;
        const float* t = txt + (size_t)tid*Dv;
        for (int d = 0; d < Dv; d++) dot += qi[d] * t[d];
        float z = dot*sc + bi;
        float lbl = (tid == i) ? 1.0f : -1.0f;
        local += neg_logsig(lbl * z) * (1.0f / (float)Bv);
    }
    for (int j = tid; j < NPv; j += 256) {
        float dot = 0.0f;
        const float* q = npf + (size_t)j*Dv;
        for (int d = 0; d < Dv; d++) dot += qi[d] * q[d];
        float z = dot*sc + bi;
        float lbl = (npidx[j] == i) ? 1.0f : -1.0f;
        local += neg_logsig(lbl * z) * (1.0f / (float)NPv);   // NPC_SCALE=1
    }
    red[tid] = local;
    __syncthreads();
    for (int s = 128; s > 0; s >>= 1) {
        if (tid < s) red[tid] += red[tid + s];
        __syncthreads();
    }
    if (tid == 0) atomicAdd(out, red[0]);
}

// ---------------- host ----------------
extern "C" void kernel_launch(void* const* d_in, const int* in_sizes, int n_in,
                              void* d_out, int out_size, void* d_ws, size_t ws_size,
                              hipStream_t stream) {
    const float* img   = (const float*)d_in[0];
    const float* txt   = (const float*)d_in[1];
    const float* scale = (const float*)d_in[2];
    const float* bias  = (const float*)d_in[3];
    const float* npf   = (const float*)d_in[4];
    const int*   npidx = (const int*)d_in[5];
    const float* tok   = (const float*)d_in[6];
    float* out = (float*)d_out;

    // workspace layout (bytes, 256-aligned)
    char* ws = (char*)d_ws;
    const long tokN = (long)Bv * Lv * Dv;       // 56,721,408
    const long npN  = (long)NPv * Dv;           // 786,432
    __bf16* tokb    = (__bf16*)ws;                                   // 113,442,816 B
    __bf16* npb     = (__bf16*)(ws + 113442816);                     // 1,572,864 B
    float*  rownorm = (float*)(ws + 113442816 + 1572864);            // 295,424 B
    float*  colmax  = (float*)(ws + 113442816 + 1572864 + 295424);   // 524,288 B
    float*  colsum  = (float*)(ws + 113442816 + 1572864 + 295424 + 524288);

    hipMemsetAsync(d_out, 0, sizeof(float), stream);

    k0_convert<<<4096, 256, 0, stream>>>(tok, tokb, tokN);
    k0_convert<<<512,  256, 0, stream>>>(npf, npb, npN);
    k1_rownorm<<<dim3(LTv, Bv),     256, 0, stream>>>(tokb, npb, rownorm);
    k2_colstats<<<dim3(NPv/16, Bv), 256, 0, stream>>>(tokb, npb, rownorm, colmax, colsum);
    k3_xac<<<dim3(NPv/16, Bv),      256, 0, stream>>>(tokb, npb, npf, rownorm, colmax, colsum,
                                                      npidx, scale, bias, out);
    k4_sig<<<Bv, 256, 0, stream>>>(img, txt, npf, npidx, scale, bias, out);
}
